// CausalSelfAttention_49237505081764
// MI455X (gfx1250) — matrix operand order repaired and checked by reading, not compiled, not run on hardware
//
#include <hip/hip_runtime.h>
#include <math.h>
#include <stdint.h>

// ---------------------------------------------------------------------------
// CDNA5 (gfx1250) causal self-attention: bf16 WMMA + async global->LDS
// staging with double buffering. wave32, 8-wave workgroups.
// ---------------------------------------------------------------------------

typedef __bf16 bf16;
typedef __attribute__((ext_vector_type(16))) __bf16 v16bf;
typedef __attribute__((ext_vector_type(8)))  __bf16 bf16x8;
typedef __attribute__((ext_vector_type(8)))  float  v8f;

#define T_SEQ 2048
#define C_EMB 768
#define N_HEADS 12
#define D_HEAD 64

// --- CDNA5 async global->LDS copy (ASYNCcnt path), with sync fallback ------
#if defined(__has_builtin)
#if __has_builtin(__builtin_amdgcn_global_load_async_to_lds_b128) && \
    __has_builtin(__builtin_amdgcn_s_wait_asynccnt)
#define USE_ASYNC_LDS 1
#endif
#endif
#ifndef USE_ASYNC_LDS
#define USE_ASYNC_LDS 0
#endif

#if USE_ASYNC_LDS
// Builtin signature (from hipcc diagnostic): args are pointers to
// "int __attribute__((vector_size(16)))" in AS1 (global) / AS3 (LDS),
// plus two integer immediates (offset, cpol).
typedef int v4i_vs __attribute__((vector_size(16)));
typedef __attribute__((address_space(1))) v4i_vs* g_v4i_ptr;
typedef __attribute__((address_space(3))) v4i_vs* l_v4i_ptr;
#endif

__device__ __forceinline__ void async_copy_b128(const bf16* gsrc, bf16* ldst) {
#if USE_ASYNC_LDS
  __builtin_amdgcn_global_load_async_to_lds_b128(
      (g_v4i_ptr)(uintptr_t)gsrc, (l_v4i_ptr)(uintptr_t)ldst,
      /*imm offset=*/0, /*cpol=*/0);
#else
  *(bf16x8*)ldst = *(const bf16x8*)gsrc;
#endif
}

template <int N>
__device__ __forceinline__ void wait_async() {
#if USE_ASYNC_LDS
  __builtin_amdgcn_s_wait_asynccnt(N);
#endif
}

// --- WMMA wrappers ----------------------------------------------------------
__device__ __forceinline__ v8f wmma_bf16(v16bf a, v16bf b, v8f c) {
  return __builtin_amdgcn_wmma_f32_16x16x32_bf16(
      /*neg_a=*/false, a, /*neg_b=*/false, b,
      /*c_mod=*/(short)0, c, /*reuse_a=*/false, /*reuse_b=*/false);
}

// A-operand fragment (16x32 bf16, M x K). Tile is [row][k], k contiguous.
// Lanes 0-15: row = lane,    K = {k0+0..7, k0+16..23}
// Lanes16-31: row = lane-16, K = {k0+8..15, k0+24..31}
__device__ __forceinline__ v16bf load_frag_a(const bf16* tile, int row_stride,
                                             int lane, int k0) {
  int hf = (lane >> 4) & 1;
  const bf16* r = tile + (size_t)(lane & 15) * row_stride;
  bf16x8 lo = *(const bf16x8*)(r + k0 + hf * 8);
  bf16x8 hi = *(const bf16x8*)(r + k0 + 16 + hf * 8);
  return __builtin_shufflevector(lo, hi, 0, 1, 2, 3, 4, 5, 6, 7,
                                 8, 9, 10, 11, 12, 13, 14, 15);
}

// B-operand fragment (32x16 bf16, K x N). Tile is [n][k], k contiguous.
// Lanes 0-15: col = lane,    K = {k0+0..7, k0+16..23}
// Lanes16-31: col = lane-16, K = {k0+8..15, k0+24..31}
__device__ __forceinline__ v16bf load_frag_b(const bf16* tile, int row_stride,
                                             int lane, int k0) {
  int hf = (lane >> 4) & 1;
  const bf16* r = tile + (size_t)(lane & 15) * row_stride;
  bf16x8 lo = *(const bf16x8*)(r + k0 + hf * 8);
  bf16x8 hi = *(const bf16x8*)(r + k0 + 16 + hf * 8);
  return __builtin_shufflevector(lo, hi, 0, 1, 2, 3, 4, 5, 6, 7,
                                 8, 9, 10, 11, 12, 13, 14, 15);
}

// ---------------------------------------------------------------------------
__global__ __launch_bounds__(256) void f32_to_bf16_kernel(
    const float* __restrict__ in, bf16* __restrict__ out, int n) {
  int i = blockIdx.x * 256 + threadIdx.x;
  if (i < n) out[i] = (bf16)in[i];
}

// in: [R][C] fp32 -> out: [C][R] bf16 (one-shot weight transpose).
__global__ __launch_bounds__(256) void f32_to_bf16_transpose_kernel(
    const float* __restrict__ in, bf16* __restrict__ out, int R, int C) {
  int i = blockIdx.x * 256 + threadIdx.x;
  if (i < R * C) {
    int r = i / C, c = i % C;
    out[(size_t)c * R + r] = (bf16)in[i];
  }
}

// ---------------------------------------------------------------------------
// Tiled GEMM: C[M,N] = A[M,K](bf16) * Bt[N,K](bf16, pre-transposed) + bias.
// Block tile 128x128, K step 32, double-buffered async LDS staging.
// 8 waves in a 2x4 grid, 4x2 WMMA tiles per wave, fp32 accumulate.
// MODE 0: fp32 C out.  MODE 1: scatter QKV -> Q[B,H,T,D], K[B,H,T,D],
// Vt[B,H,D,T] as bf16.
template <int MODE>
__global__ __launch_bounds__(256) void gemm_bf16_kernel(
    const bf16* __restrict__ A, const bf16* __restrict__ Bt,
    const float* __restrict__ bias, float* __restrict__ Cout,
    bf16* __restrict__ Qb, bf16* __restrict__ Kb, bf16* __restrict__ Vtb,
    int M, int N, int K) {
  __shared__ bf16 As[2][128][32];   // [buf][m][k]
  __shared__ bf16 Bs[2][128][32];   // [buf][n][k]

  const int tid  = threadIdx.x;
  const int lane = tid & 31;
  const int wave = tid >> 5;
  const int wm = wave & 1;          // 2 x 64 rows
  const int wn = wave >> 1;         // 4 x 32 cols
  const int bm = blockIdx.x * 128;
  const int bn = blockIdx.y * 128;

  // 1024 contiguous 16B chunks per tile pair; 4 async copies per thread.
  auto stage = [&](int buf, int k0) {
    #pragma unroll
    for (int i = 0; i < 2; ++i) {
      int idx = tid + i * 256;
      int row = idx >> 2;            // 0..127
      int kc  = (idx & 3) * 8;       // 0,8,16,24
      async_copy_b128(A  + (size_t)(bm + row) * K + k0 + kc, &As[buf][row][kc]);
      async_copy_b128(Bt + (size_t)(bn + row) * K + k0 + kc, &Bs[buf][row][kc]);
    }
  };

  v8f acc[4][2] = {};
  const int nk = K / 32;

  stage(0, 0);
  for (int kt = 0; kt < nk; ++kt) {
    const int cur = kt & 1;
    if (kt + 1 < nk) {
      stage(cur ^ 1, (kt + 1) * 32);   // prefetch next tile (async)
      wait_async<4>();                 // current tile resident, next in flight
    } else {
      wait_async<0>();
    }
    __syncthreads();

    // Batch all fragment loads first so the WMMA chain doesn't stall on
    // per-instruction LDS drains.
    v16bf bfrag[2], afrag[4];
    #pragma unroll
    for (int j = 0; j < 2; ++j)
      bfrag[j] = load_frag_b(&Bs[cur][wn * 32 + j * 16][0], 32, lane, 0);
    #pragma unroll
    for (int i = 0; i < 4; ++i)
      afrag[i] = load_frag_a(&As[cur][wm * 64 + i * 16][0], 32, lane, 0);
    #pragma unroll
    for (int i = 0; i < 4; ++i)
      #pragma unroll
      for (int j = 0; j < 2; ++j)
        acc[i][j] = wmma_bf16(afrag[i], bfrag[j], acc[i][j]);
    __syncthreads();
  }

  // Epilogue. C/D layout: VGPR r -> row (r + 8*half), col = lane&15.
  const int hf  = (lane >> 4) & 1;
  const int ncl = lane & 15;
  #pragma unroll
  for (int i = 0; i < 4; ++i) {
    #pragma unroll
    for (int j = 0; j < 2; ++j) {
      int mbase = bm + wm * 64 + i * 16 + hf * 8;
      int ncol  = bn + wn * 32 + j * 16 + ncl;
      float bv = bias[ncol];
      #pragma unroll
      for (int r = 0; r < 8; ++r) {
        float val = acc[i][j][r] + bv;
        int m = mbase + r;
        if (MODE == 0) {
          Cout[(size_t)m * N + ncol] = val;
        } else {
          int b = m / T_SEQ, t = m % T_SEQ;
          int which = ncol / C_EMB;
          int c = ncol % C_EMB;
          int h = c / D_HEAD, d = c % D_HEAD;
          size_t bh = (size_t)b * N_HEADS + h;
          if (which == 0)
            Qb[(bh * T_SEQ + t) * D_HEAD + d] = (bf16)val;
          else if (which == 1)
            Kb[(bh * T_SEQ + t) * D_HEAD + d] = (bf16)val;
          else
            Vtb[(bh * D_HEAD + d) * T_SEQ + t] = (bf16)val;
        }
      }
    }
  }
}

// ---------------------------------------------------------------------------
// Flash attention: one workgroup per (b*h, 128-query block). 8 waves, each
// owns 16 query rows. 64-key blocks double-buffered in LDS via async copies;
// online softmax; O accumulated in fp32 WMMA tiles.
__global__ __launch_bounds__(256) void flash_attn_kernel(
    const bf16* __restrict__ Qb, const bf16* __restrict__ Kb,
    const bf16* __restrict__ Vtb, bf16* __restrict__ Yb) {
  __shared__ bf16 Ks[2][64][D_HEAD];    // [buf][key][d]
  __shared__ bf16 Vs[2][D_HEAD][64];    // [buf][d][key] (V transposed)
  __shared__ bf16 Ps[8][16][64];        // per-wave P tile [qrow][key]

  const int tid  = threadIdx.x;
  const int lane = tid & 31;
  const int wave = tid >> 5;
  const int hf   = (lane >> 4) & 1;
  const int ncl  = lane & 15;

  const int bh   = blockIdx.x;          // 0..23
  const int qblk = blockIdx.y;          // 0..15
  const int b    = bh / N_HEADS;
  const int h    = bh % N_HEADS;

  const bf16* Qh = Qb  + (size_t)bh * T_SEQ * D_HEAD;
  const bf16* Kh = Kb  + (size_t)bh * T_SEQ * D_HEAD;
  const bf16* Vh = Vtb + (size_t)bh * D_HEAD * T_SEQ;

  const int qbase = qblk * 128;
  const int qrow0 = qbase + wave * 16;

  auto stage_kv = [&](int buf, int kb) {
    int kt0 = kb * 64;
    #pragma unroll
    for (int i = 0; i < 2; ++i) {
      int idx = tid + i * 256;
      int row = idx >> 3;              // 0..63
      int cc  = (idx & 7) * 8;         // 0..56
      async_copy_b128(Kh + (size_t)(kt0 + row) * D_HEAD + cc,
                      &Ks[buf][row][cc]);
      async_copy_b128(Vh + (size_t)row * T_SEQ + kt0 + cc,
                      &Vs[buf][row][cc]);
    }
  };

  // Loop-invariant Q fragments (16 rows x D_HEAD, two K=32 chunks).
  v16bf qfrag0 = load_frag_a(Qh + (size_t)qrow0 * D_HEAD, D_HEAD, lane, 0);
  v16bf qfrag1 = load_frag_a(Qh + (size_t)qrow0 * D_HEAD, D_HEAD, lane, 32);

  v8f o[4] = {};
  float m_i[8], l_i[8];
  #pragma unroll
  for (int r = 0; r < 8; ++r) { m_i[r] = -1e30f; l_i[r] = 0.0f; }

  const float scale = 0.125f;           // 1/sqrt(64)
  const int nkb = qblk * 2 + 2;         // cover keys 0 .. qbase+127

  stage_kv(0, 0);
  for (int kb = 0; kb < nkb; ++kb) {
    const int cur = kb & 1;
    const int kt0 = kb * 64;
    if (kb + 1 < nkb) {
      stage_kv(cur ^ 1, kb + 1);
      wait_async<4>();
    } else {
      wait_async<0>();
    }
    __syncthreads();

    // S = Q * K^T : batch all 8 K-fragment loads, then 8 WMMAs back-to-back.
    v16bf kfrag[4][2];
    #pragma unroll
    for (int nt = 0; nt < 4; ++nt) {
      kfrag[nt][0] = load_frag_b(&Ks[cur][nt * 16][0], D_HEAD, lane, 0);
      kfrag[nt][1] = load_frag_b(&Ks[cur][nt * 16][0], D_HEAD, lane, 32);
    }
    v8f s[4];
    #pragma unroll
    for (int nt = 0; nt < 4; ++nt) {
      v8f a = {};
      a = wmma_bf16(qfrag0, kfrag[nt][0], a);
      a = wmma_bf16(qfrag1, kfrag[nt][1], a);
      s[nt] = a;
    }

    // Scale + causal mask.
    #pragma unroll
    for (int nt = 0; nt < 4; ++nt) {
      int kg = kt0 + nt * 16 + ncl;
      #pragma unroll
      for (int r = 0; r < 8; ++r) {
        int qg = qrow0 + hf * 8 + r;
        float v = s[nt][r] * scale;
        s[nt][r] = (kg <= qg) ? v : -1e30f;
      }
    }

    // Online softmax (reductions within each 16-lane half of the wave).
    #pragma unroll
    for (int r = 0; r < 8; ++r) {
      float rm = fmaxf(fmaxf(s[0][r], s[1][r]), fmaxf(s[2][r], s[3][r]));
      #pragma unroll
      for (int off = 1; off < 16; off <<= 1)
        rm = fmaxf(rm, __shfl_xor(rm, off, 32));
      float mnew = fmaxf(m_i[r], rm);
      float corr = __expf(m_i[r] - mnew);
      m_i[r] = mnew;
      float lsum = 0.0f;
      #pragma unroll
      for (int nt = 0; nt < 4; ++nt) {
        float p = __expf(s[nt][r] - mnew);
        s[nt][r] = p;
        lsum += p;
      }
      #pragma unroll
      for (int off = 1; off < 16; off <<= 1)
        lsum += __shfl_xor(lsum, off, 32);
      l_i[r] = l_i[r] * corr + lsum;
      #pragma unroll
      for (int dt = 0; dt < 4; ++dt) o[dt][r] *= corr;
    }

    // Repack P (C-layout regs) through per-wave LDS into A-fragment layout.
    #pragma unroll
    for (int nt = 0; nt < 4; ++nt)
      #pragma unroll
      for (int r = 0; r < 8; ++r)
        Ps[wave][hf * 8 + r][nt * 16 + ncl] = (bf16)s[nt][r];

    // Batch P and V fragment loads, then run the 8 O-update WMMAs.
    v16bf pfrag0 = load_frag_a(&Ps[wave][0][0], 64, lane, 0);
    v16bf pfrag1 = load_frag_a(&Ps[wave][0][0], 64, lane, 32);
    v16bf vfrag[4][2];
    #pragma unroll
    for (int dt = 0; dt < 4; ++dt) {
      vfrag[dt][0] = load_frag_b(&Vs[cur][dt * 16][0], 64, lane, 0);
      vfrag[dt][1] = load_frag_b(&Vs[cur][dt * 16][0], 64, lane, 32);
    }
    #pragma unroll
    for (int dt = 0; dt < 4; ++dt) {
      o[dt] = wmma_bf16(pfrag0, vfrag[dt][0], o[dt]);
      o[dt] = wmma_bf16(pfrag1, vfrag[dt][1], o[dt]);
    }
    __syncthreads();   // all waves done with Ks/Vs[cur] before restaging it
  }

  // Finalize and scatter into y[b, t, h*64+d] (bf16 for projection GEMM).
  #pragma unroll
  for (int r = 0; r < 8; ++r) {
    float inv = 1.0f / l_i[r];
    #pragma unroll
    for (int dt = 0; dt < 4; ++dt) o[dt][r] *= inv;
  }
  #pragma unroll
  for (int dt = 0; dt < 4; ++dt) {
    #pragma unroll
    for (int r = 0; r < 8; ++r) {
      int t = qrow0 + hf * 8 + r;
      int d = dt * 16 + ncl;
      Yb[((size_t)b * T_SEQ + t) * C_EMB + h * D_HEAD + d] = (bf16)o[dt][r];
    }
  }
}

// ---------------------------------------------------------------------------
extern "C" void kernel_launch(void* const* d_in, const int* in_sizes, int n_in,
                              void* d_out, int out_size, void* d_ws, size_t ws_size,
                              hipStream_t stream) {
  const float* x      = (const float*)d_in[0];  // [2,2048,768]
  const float* W_attn = (const float*)d_in[1];  // [768,2304]
  const float* b_attn = (const float*)d_in[2];  // [2304]
  const float* W_proj = (const float*)d_in[3];  // [768,768]
  const float* b_proj = (const float*)d_in[4];  // [768]
  float* out = (float*)d_out;                   // [2,2048,768]

  const int B = 2;
  const int M = B * T_SEQ;                      // 4096
  const int N_QKV = 3 * C_EMB;                  // 2304

  // Workspace carve-up (bf16 buffers, 256B aligned).
  char* ws = (char*)d_ws;
  auto take = [&](size_t elems) {
    bf16* p = (bf16*)ws;
    size_t bytes = (elems * sizeof(bf16) + 255) & ~(size_t)255;
    ws += bytes;
    return p;
  };
  bf16* xb    = take((size_t)M * C_EMB);
  bf16* WabT  = take((size_t)C_EMB * N_QKV);    // [2304][768]
  bf16* WpbT  = take((size_t)C_EMB * C_EMB);    // [768][768] transposed
  size_t qkv_elems = (size_t)B * N_HEADS * T_SEQ * D_HEAD;
  bf16* Qb  = take(qkv_elems);
  bf16* Kb  = take(qkv_elems);
  bf16* Vtb = take(qkv_elems);
  bf16* Yb  = take((size_t)M * C_EMB);

  // 1) fp32 -> bf16 conversions; weights transposed once to [N][K].
  {
    int n1 = M * C_EMB;
    f32_to_bf16_kernel<<<(n1 + 255) / 256, 256, 0, stream>>>(x, xb, n1);
    int n2 = C_EMB * N_QKV;
    f32_to_bf16_transpose_kernel<<<(n2 + 255) / 256, 256, 0, stream>>>(
        W_attn, WabT, C_EMB, N_QKV);
    int n3 = C_EMB * C_EMB;
    f32_to_bf16_transpose_kernel<<<(n3 + 255) / 256, 256, 0, stream>>>(
        W_proj, WpbT, C_EMB, C_EMB);
  }

  // 2) QKV GEMM: [4096,768] x [768,2304], scatter to Q/K/Vt (bf16).
  gemm_bf16_kernel<1><<<dim3(M / 128, N_QKV / 128), 256, 0, stream>>>(
      xb, WabT, b_attn, nullptr, Qb, Kb, Vtb, M, N_QKV, C_EMB);

  // 3) Flash attention over 24 (b,h) heads x 16 query blocks.
  flash_attn_kernel<<<dim3(B * N_HEADS, T_SEQ / 128), 256, 0, stream>>>(
      Qb, Kb, Vtb, Yb);

  // 4) Output projection: [4096,768] x [768,768] + bias -> fp32 out.
  gemm_bf16_kernel<0><<<dim3(M / 128, C_EMB / 128), 256, 0, stream>>>(
      Yb, WpbT, b_proj, out, nullptr, nullptr, nullptr, M, C_EMB, C_EMB);
}
